// RewaHierarchicalAttention_90237262889105
// MI455X (gfx1250) — matrix operand order repaired and checked by reading, not compiled, not run on hardware
//
#include <hip/hip_runtime.h>
#include <cstdint>
#include <cstddef>

#define B_ 4
#define N_ 4096
#define E_ 1024
#define H_ 16
#define DH_ 64

typedef __bf16 bf16_t;
typedef __attribute__((ext_vector_type(16))) __bf16 v16bf;
typedef __attribute__((ext_vector_type(8)))  __bf16 v8bf;
typedef __attribute__((ext_vector_type(8)))  float  v8f;

static __device__ __forceinline__ v8f wmma_bf16(v16bf a, v16bf b, v8f c) {
  // D = A(16x32 bf16) * B(32x16 bf16) + C(16x16 f32)
  return __builtin_amdgcn_wmma_f32_16x16x32_bf16(false, a, false, b, (short)0, c,
                                                 false, false);
}

static __device__ __forceinline__ float rowmax16(float v) {
#pragma unroll
  for (int off = 1; off < 16; off <<= 1) v = fmaxf(v, __shfl_xor(v, off, 16));
  return v;
}
static __device__ __forceinline__ float rowsum16(float v) {
#pragma unroll
  for (int off = 1; off < 16; off <<= 1) v += __shfl_xor(v, off, 16);
  return v;
}

// ---------------------------------------------------------------------------
// fp32 -> bf16 (with scale); also used for the /3 level average
// ---------------------------------------------------------------------------
__global__ void rewa_cvt_kernel(const float* __restrict__ in,
                                bf16_t* __restrict__ out, size_t n, float scale) {
  size_t i = (size_t)blockIdx.x * blockDim.x + threadIdx.x;
  const size_t st = (size_t)gridDim.x * blockDim.x;
  for (; i < n; i += st) out[i] = (bf16_t)(in[i] * scale);
}

__global__ void rewa_zero_kernel(float* __restrict__ p, size_t n) {
  size_t i = (size_t)blockIdx.x * blockDim.x + threadIdx.x;
  const size_t st = (size_t)gridDim.x * blockDim.x;
  for (; i < n; i += st) p[i] = 0.0f;
}

// ---------------------------------------------------------------------------
// Stable argsort (rank-based counting) per batch row; matches jnp.argsort.
// dst[rank_i] = i  so that gather(dst) yields ascending stable order.
// ---------------------------------------------------------------------------
__global__ __launch_bounds__(256) void rewa_sort_kernel(const int* __restrict__ wb,
                                                        int* __restrict__ sidx) {
  __shared__ int keys[N_];
  const int b = blockIdx.x;
  const int* src = wb + (size_t)b * N_;
  int* dst = sidx + (size_t)b * N_;
  for (int i = threadIdx.x; i < N_; i += blockDim.x) keys[i] = src[i];
  __syncthreads();
  for (int i = threadIdx.x; i < N_; i += blockDim.x) {
    const int v = keys[i];
    int rank = 0;
    for (int j = 0; j < N_; ++j) {
      const int u = keys[j];
      rank += (int)((u < v) | ((u == v) & (j < i)));
    }
    dst[rank] = i;
  }
}

// ---------------------------------------------------------------------------
// GEMM: out[M,Nn] = A[M,K](bf16) @ W[Nn,K]^T (bf16) + bias.
// One wave computes a 32x64 macro-tile (2 M-tiles x 4 N-tiles): per K=32 step
// it loads 2 A-frags + 4 B-frags (12 x b128) and issues 8 independent WMMAs.
// Per-lane fragment addressing follows the CDNA5 ISA layouts:
//   A lane: row = lane%16; K-runs [half*8,+8) and [16+half*8,+8) per 32-step
//   B lane: col = lane%16; K-runs [half*8,+8) and [16+half*8,+8) per 32-step (contig in W row)
//   D lane: col = lane%16; rows r+8*half for r=0..7
// ---------------------------------------------------------------------------
__global__ __launch_bounds__(256) void rewa_gemm_kernel(
    const bf16_t* __restrict__ A, const bf16_t* __restrict__ W,
    const float* __restrict__ bias, float* __restrict__ outF,
    bf16_t* __restrict__ outB, int M, int K, int Nn, int storeF) {
  const int lane = threadIdx.x & 31;
  const int wave = threadIdx.x >> 5;
  const int mt = blockIdx.x * 8 + wave;     // macro-tile id
  const int mtM = M >> 5;                   // macro rows of 32
  const int mtN = Nn >> 6;                  // macro cols of 64
  if (mt >= mtM * mtN) return;
  const int tm = (mt % mtM) * 2;            // base 16-row tile index
  const int tn = (mt / mtM) * 4;            // base 16-col tile index
  const int half = lane >> 4;
  const int l16 = lane & 15;

  const bf16_t* ap0 = A + (size_t)(tm * 16 + l16) * K + half * 8;
  const bf16_t* ap1 = ap0 + (size_t)16 * K;
  const bf16_t* wp = W + (size_t)(tn * 16 + l16) * K + half * 8;
  const size_t wstep = (size_t)16 * K;

  v8f c[2][4] = {};
  for (int kk = 0; kk < K; kk += 32) {
    __builtin_prefetch(ap0 + kk + 512, 0, 0);
    __builtin_prefetch(ap1 + kk + 512, 0, 0);
    v16bf a[2], w4[4];
    {
      v8bf l0 = *(const v8bf*)(ap0 + kk);
      v8bf h0 = *(const v8bf*)(ap0 + kk + 16);
      v8bf l1 = *(const v8bf*)(ap1 + kk);
      v8bf h1 = *(const v8bf*)(ap1 + kk + 16);
#pragma unroll
      for (int j = 0; j < 8; ++j) {
        a[0][j] = l0[j]; a[0][j + 8] = h0[j];
        a[1][j] = l1[j]; a[1][j + 8] = h1[j];
      }
    }
#pragma unroll
    for (int j4 = 0; j4 < 4; ++j4) {
      v8bf bl = *(const v8bf*)(wp + j4 * wstep + kk);
      v8bf bh = *(const v8bf*)(wp + j4 * wstep + kk + 16);
#pragma unroll
      for (int j = 0; j < 8; ++j) { w4[j4][j] = bl[j]; w4[j4][j + 8] = bh[j]; }
    }
#pragma unroll
    for (int i = 0; i < 2; ++i)
#pragma unroll
      for (int j4 = 0; j4 < 4; ++j4)
        c[i][j4] = wmma_bf16(a[i], w4[j4], c[i][j4]);
  }

#pragma unroll
  for (int j4 = 0; j4 < 4; ++j4) {
    const float bc = bias[(tn + j4) * 16 + l16];
#pragma unroll
    for (int i = 0; i < 2; ++i) {
#pragma unroll
      for (int r = 0; r < 8; ++r) {
        const int row = (tm + i) * 16 + r + half * 8;
        const size_t o = (size_t)row * Nn + (tn + j4) * 16 + l16;
        const float v = c[i][j4][r] + bc;
        if (storeF) outF[o] = v; else outB[o] = (bf16_t)v;
      }
    }
  }
}

// ---------------------------------------------------------------------------
// Chunked attention, flash-style. One wave = one 16-query slab of one chunk
// of one (batch,head). Keys = [prev chunk | cur chunk] (2C rows, D=64),
// processed 32 keys per iteration so P@V uses full K=32 WMMA fragments.
// V tiles are staged into per-wave LDS with global_load_async_to_lds_b128
// (ASYNCcnt); P goes through LDS for the D->A layout flip (DScnt).
// Scatter-adds (race-free within a level) into the f32 accumulator at the
// unsorted row positions.
// ---------------------------------------------------------------------------
__global__ __launch_bounds__(256) void rewa_attn_kernel(
    const bf16_t* __restrict__ Q, const bf16_t* __restrict__ Kb,
    const bf16_t* __restrict__ Vb, const int* __restrict__ sidx,
    float* __restrict__ accum, int C) {
  __shared__ __align__(16) bf16_t Plds[8][16][32];   // per-wave P (16x32)
  __shared__ __align__(16) bf16_t Vt[8][32][64];     // per-wave V pair-tile

  const int lane = threadIdx.x & 31;
  const int wave = threadIdx.x >> 5;
  const int S = blockIdx.x * 8 + wave;   // global slab id, BH*(N/16) = 16384
  const int bh = S >> 8;                 // 256 slabs per (batch,head)
  const int sin = S & 255;
  const int b = bh >> 4;
  const int h = bh & 15;
  const int cs = C >> 4;                 // slabs per chunk
  const int ci = sin / cs;
  const int sc = sin - ci * cs;
  const int half = lane >> 4;
  const int l16 = lane & 15;
  const int qbase = ci * C + sc * 16;
  const int* sx = sidx + (size_t)b * N_;

  // ---- Q fragments: 16 rows x 64 dims -> two K=32 A-frags ----
  v16bf a0, a1;
  {
    const int qrow = sx[qbase + l16];
    const bf16_t* qp = Q + ((size_t)b * N_ + qrow) * E_ + h * DH_;
    v8bf lo0 = *(const v8bf*)(qp + half * 8);
    v8bf hi0 = *(const v8bf*)(qp + 16 + half * 8);
    v8bf lo1 = *(const v8bf*)(qp + 32 + half * 8);
    v8bf hi1 = *(const v8bf*)(qp + 48 + half * 8);
#pragma unroll
    for (int j = 0; j < 8; ++j) {
      a0[j] = lo0[j]; a0[j + 8] = hi0[j];
      a1[j] = lo1[j]; a1[j + 8] = hi1[j];
    }
  }

  float mrow[8], lrow[8];
  v8f acc[4] = {};
#pragma unroll
  for (int r = 0; r < 8; ++r) { mrow[r] = -3.0e38f; lrow[r] = 0.0f; }

  const int npairs = C >> 4;             // (2C)/32 iterations of 32 keys
  for (int t2 = 0; t2 < npairs; ++t2) {
    // ---- async-stage V pair-tile: lane -> gathered key row (32x64 bf16) ----
    {
      const int vk = t2 * 32 + lane;
      const bool vprev = vk < C;
      int vpos = vprev ? ((ci == 0) ? vk : ((ci - 1) * C + vk))
                       : (ci * C + vk - C);
      const int vrow = sx[vpos];
      const bf16_t* vp = Vb + ((size_t)b * N_ + vrow) * E_ + h * DH_;
      const unsigned ldso = (unsigned)(uintptr_t)&Vt[wave][lane][0];
      // s_wait_dscnt guards WAR vs previous iteration's ds_loads of Vt.
      asm volatile(
          "s_wait_dscnt 0x0\n\t"
          "global_load_async_to_lds_b128 %0, %1, off\n\t"
          "global_load_async_to_lds_b128 %0, %1, off offset:16\n\t"
          "global_load_async_to_lds_b128 %0, %1, off offset:32\n\t"
          "global_load_async_to_lds_b128 %0, %1, off offset:48\n\t"
          "global_load_async_to_lds_b128 %0, %1, off offset:64\n\t"
          "global_load_async_to_lds_b128 %0, %1, off offset:80\n\t"
          "global_load_async_to_lds_b128 %0, %1, off offset:96\n\t"
          "global_load_async_to_lds_b128 %0, %1, off offset:112"
          :: "v"(ldso), "v"(vp) : "memory");
    }

    // ---- scores for both 16-key sub-tiles: S = Q @ K^T (K=64 chained) ----
    v8f s[2];
    bool maskd[2];
#pragma unroll
    for (int u = 0; u < 2; ++u) {
      const int kk = t2 * 32 + u * 16 + l16;  // concat index of score column
      const bool prevh = kk < C;
      maskd[u] = prevh && (ci == 0);
      int kpos = prevh ? ((ci - 1) * C + kk) : (ci * C + kk - C);
      if (maskd[u]) kpos = kk;                // safe address; masked below
      const int krow = sx[kpos];
      const bf16_t* kp = Kb + ((size_t)b * N_ + krow) * E_ + h * DH_;
      v16bf b0, b1;
      {
        v8bf p0 = *(const v8bf*)(kp + half * 8);
        v8bf p1 = *(const v8bf*)(kp + 16 + half * 8);
        v8bf p2 = *(const v8bf*)(kp + 32 + half * 8);
        v8bf p3 = *(const v8bf*)(kp + 48 + half * 8);
#pragma unroll
        for (int j = 0; j < 8; ++j) {
          b0[j] = p0[j]; b0[j + 8] = p1[j];
          b1[j] = p2[j]; b1[j + 8] = p3[j];
        }
      }
      v8f sv = v8f{};
      sv = wmma_bf16(a0, b0, sv);
      sv = wmma_bf16(a1, b1, sv);
      s[u] = sv;
    }

    // ---- online softmax over 32 keys (per-row stats via 16-lane groups) ----
    float p0[8], p1[8], rescale[8];
#pragma unroll
    for (int r = 0; r < 8; ++r) {
      const float sr0 = maskd[0] ? -1.0e9f : s[0][r] * 0.125f;  // 1/sqrt(64)
      const float sr1 = maskd[1] ? -1.0e9f : s[1][r] * 0.125f;
      const float tmx = rowmax16(fmaxf(sr0, sr1));
      const float mn = fmaxf(mrow[r], tmx);
      const float e0 = __expf(sr0 - mn);
      const float e1 = __expf(sr1 - mn);
      const float ts = rowsum16(e0 + e1);
      rescale[r] = __expf(mrow[r] - mn);
      lrow[r] = lrow[r] * rescale[r] + ts;
      mrow[r] = mn;
      p0[r] = e0; p1[r] = e1;
    }
#pragma unroll
    for (int dt = 0; dt < 4; ++dt) {
#pragma unroll
      for (int r = 0; r < 8; ++r) acc[dt][r] *= rescale[r];
    }

    // ---- P: D-layout -> 16x32 A-layout through per-wave LDS ----
#pragma unroll
    for (int r = 0; r < 8; ++r) {
      Plds[wave][r + 8 * half][l16] = (bf16_t)p0[r];
      Plds[wave][r + 8 * half][16 + l16] = (bf16_t)p1[r];
    }
    // P visible (DScnt) and V pair-tile landed (ASYNCcnt) before reads.
    asm volatile("s_wait_dscnt 0x0\n\t"
                 "s_wait_asynccnt 0x0" ::: "memory");
    v16bf pf;
    {
      v8bf plo = *(const v8bf*)&Plds[wave][l16][half * 8];
      v8bf phi = *(const v8bf*)&Plds[wave][l16][16 + half * 8];
#pragma unroll
      for (int j = 0; j < 8; ++j) { pf[j] = plo[j]; pf[j + 8] = phi[j]; }
    }

    // ---- O += P(16x32) @ V(32x64): full K=32 fragments ----
#pragma unroll
    for (int dt = 0; dt < 4; ++dt) {
      v16bf bvv;
#pragma unroll
      for (int j = 0; j < 16; ++j)
        bvv[j] = Vt[wave][(j < 8 ? half * 8 + j : 8 + half * 8 + j)][dt * 16 + l16];
      acc[dt] = wmma_bf16(pf, bvv, acc[dt]);
    }
  }

  // ---- normalize and scatter-accumulate at unsorted positions ----
#pragma unroll
  for (int r = 0; r < 8; ++r) {
    const float inv = 1.0f / lrow[r];
    const int row = sx[qbase + r + 8 * half];
    float* op = accum + ((size_t)b * N_ + row) * E_ + h * DH_ + l16;
    op[0]  += acc[0][r] * inv;
    op[16] += acc[1][r] * inv;
    op[32] += acc[2][r] * inv;
    op[48] += acc[3][r] * inv;
  }
}

// ---------------------------------------------------------------------------
extern "C" void kernel_launch(void* const* d_in, const int* in_sizes, int n_in,
                              void* d_out, int out_size, void* d_ws, size_t ws_size,
                              hipStream_t stream) {
  const float* x  = (const float*)d_in[0];
  const float* Wq = (const float*)d_in[1];
  const float* bq = (const float*)d_in[2];
  const float* Wk = (const float*)d_in[3];
  const float* bk = (const float*)d_in[4];
  const float* Wv = (const float*)d_in[5];
  const float* bv = (const float*)d_in[6];
  const float* Wo = (const float*)d_in[7];
  const float* bo = (const float*)d_in[8];
  const int* wbc = (const int*)d_in[9];
  const int* wbm = (const int*)d_in[10];
  const int* wbf = (const int*)d_in[11];

  const int M = B_ * N_, K = E_, Nn = E_;
  const size_t nxe = (size_t)M * E_;     // 16.7M elements
  const size_t nww = (size_t)E_ * E_;

  char* w = (char*)d_ws;
  auto take = [&](size_t bytes) {
    char* p = w;
    w += (bytes + 255) & ~(size_t)255;
    return p;
  };
  bf16_t* xb   = (bf16_t*)take(nxe * 2);
  bf16_t* qb   = (bf16_t*)take(nxe * 2);
  bf16_t* kb   = (bf16_t*)take(nxe * 2);
  bf16_t* vb   = (bf16_t*)take(nxe * 2);
  bf16_t* ab   = (bf16_t*)take(nxe * 2);
  bf16_t* wqb  = (bf16_t*)take(nww * 2);
  bf16_t* wkb  = (bf16_t*)take(nww * 2);
  bf16_t* wvb  = (bf16_t*)take(nww * 2);
  bf16_t* wob  = (bf16_t*)take(nww * 2);
  int*    sidx = (int*)take((size_t)3 * B_ * N_ * sizeof(int));
  float*  acc  = (float*)take(nxe * 4);

  dim3 blk(256);

  // bf16 copies of activations + weights
  rewa_cvt_kernel<<<2048, blk, 0, stream>>>(x, xb, nxe, 1.0f);
  rewa_cvt_kernel<<<512, blk, 0, stream>>>(Wq, wqb, nww, 1.0f);
  rewa_cvt_kernel<<<512, blk, 0, stream>>>(Wk, wkb, nww, 1.0f);
  rewa_cvt_kernel<<<512, blk, 0, stream>>>(Wv, wvb, nww, 1.0f);
  rewa_cvt_kernel<<<512, blk, 0, stream>>>(Wo, wob, nww, 1.0f);

  // q/k/v projections (WMMA GEMM, bf16 outputs feed attention)
  const int gblocks = (M / 32) * (Nn / 64) / 8;   // 32x64 macro-tile per wave
  rewa_gemm_kernel<<<gblocks, blk, 0, stream>>>(xb, wqb, bq, nullptr, qb, M, K, Nn, 0);
  rewa_gemm_kernel<<<gblocks, blk, 0, stream>>>(xb, wkb, bk, nullptr, kb, M, K, Nn, 0);
  rewa_gemm_kernel<<<gblocks, blk, 0, stream>>>(xb, wvb, bv, nullptr, vb, M, K, Nn, 0);

  // stable argsort of bucket ids, one per (level, batch)
  rewa_sort_kernel<<<B_, blk, 0, stream>>>(wbc, sidx);
  rewa_sort_kernel<<<B_, blk, 0, stream>>>(wbm, sidx + B_ * N_);
  rewa_sort_kernel<<<B_, blk, 0, stream>>>(wbf, sidx + 2 * B_ * N_);

  // attention accumulator
  rewa_zero_kernel<<<2048, blk, 0, stream>>>(acc, nxe);
  rewa_attn_kernel<<<2048, blk, 0, stream>>>(qb, kb, vb, sidx, acc, 256);
  rewa_attn_kernel<<<2048, blk, 0, stream>>>(qb, kb, vb, sidx + B_ * N_, acc, 64);
  rewa_attn_kernel<<<2048, blk, 0, stream>>>(qb, kb, vb, sidx + 2 * B_ * N_, acc, 16);

  // /3 average, then output projection into f32 d_out
  rewa_cvt_kernel<<<2048, blk, 0, stream>>>(acc, ab, nxe, 1.0f / 3.0f);
  rewa_gemm_kernel<<<gblocks, blk, 0, stream>>>(ab, wob, bo, (float*)d_out, nullptr,
                                                M, K, Nn, 1);
}
